// DNADeBruijnClassifier2_73632919323009
// MI455X (gfx1250) — matrix operand order repaired and checked by reading, not compiled, not run on hardware
//
#include <hip/hip_runtime.h>

// ---------------------------------------------------------------------------
// Types for CDNA5 WMMA
// ---------------------------------------------------------------------------
typedef __bf16 v16bf __attribute__((ext_vector_type(16)));
typedef float  v8f   __attribute__((ext_vector_type(8)));
typedef int    i32x4 __attribute__((ext_vector_type(4)));

#define G_  32
#define N_  8192
#define E_  16384

// ---------------------------------------------------------------------------
// CDNA5 async-copy helpers
// ---------------------------------------------------------------------------
template<int N>
__device__ __forceinline__ void wait_asynccnt()
{
#if __has_builtin(__builtin_amdgcn_s_wait_asynccnt)
    __builtin_amdgcn_s_wait_asynccnt(N);
#else
    asm volatile("s_wait_asynccnt %0" :: "n"(N) : "memory");
#endif
}

// one 16-byte async Global -> LDS copy per lane (wave-wide instruction,
// tracked by ASYNCcnt)
__device__ __forceinline__ void async_load_b128(unsigned lds_byte_off,
                                                const void* gptr)
{
    asm volatile("global_load_async_to_lds_b128 %0, %1, off"
                 :: "v"(lds_byte_off),
                    "v"((unsigned long long)(size_t)gptr)
                 : "memory");
}

// ---------------------------------------------------------------------------
// Weight conversion f32 -> bf16 with row/col zero padding
// ---------------------------------------------------------------------------
__global__ void cvt_w_kernel(const float* __restrict__ src, __bf16* __restrict__ dst,
                             int rows, int cols, int rowsPad, int colsPad)
{
    long long t = (long long)blockIdx.x * blockDim.x + threadIdx.x;
    long long total = (long long)rowsPad * colsPad;
    if (t >= total) return;
    int r = (int)(t / colsPad);
    int c = (int)(t % colsPad);
    float v = (r < rows && c < cols) ? src[(long long)r * cols + c] : 0.f;
    dst[t] = (__bf16)v;
}

// ---------------------------------------------------------------------------
// GINE message: msg = relu(x[src] + lin_w @ ea + lin_b), scatter-add to agg[dst]
// ---------------------------------------------------------------------------
__global__ __launch_bounds__(256)
void scatter_gine_kernel(const float* __restrict__ x, int ldx, long long strideXg,
                         const int* __restrict__ edges,   // [G,2,E]
                         const float* __restrict__ ea,    // [G,E,8]
                         const float* __restrict__ lw,    // [din,8]
                         const float* __restrict__ lb,    // [din]
                         float* __restrict__ agg,         // [G,N,128]
                         int din)
{
    int g = blockIdx.z;
    long long t = (long long)blockIdx.x * blockDim.x + threadIdx.x;
    int e = (int)(t / din);
    int c = (int)(t % din);
    if (e >= E_) return;
    const int* ep = edges + (long long)g * 2 * E_;
    int src = ep[e];
    int dst = ep[E_ + e];
    const float* eap = ea + ((long long)g * E_ + e) * 8;
    const float* lwp = lw + c * 8;
    float s = lb[c];
#pragma unroll
    for (int j = 0; j < 8; ++j) s += lwp[j] * eap[j];
    float xs = x[(long long)g * strideXg + (long long)src * ldx + c];
    float m = xs + s;
    if (m > 0.f) {
        unsafeAtomicAdd(&agg[((long long)g * N_ + dst) * 128 + c], m);
    }
}

// ---------------------------------------------------------------------------
// hin = bf16(x + agg), K-padded to kp with zeros (layer 0: din=16 -> kp=32)
// ---------------------------------------------------------------------------
__global__ __launch_bounds__(256)
void prep_hin_kernel(const float* __restrict__ x, int ldx, long long strideXg,
                     const float* __restrict__ agg,
                     __bf16* __restrict__ hin, int kp, int din)
{
    int g = blockIdx.z;
    long long t = (long long)blockIdx.x * blockDim.x + threadIdx.x;
    int n = (int)(t / kp);
    int c = (int)(t % kp);
    if (n >= N_) return;
    float v = 0.f;
    if (c < din)
        v = x[(long long)g * strideXg + (long long)n * ldx + c]
          + agg[((long long)g * N_ + n) * 128 + c];
    hin[((long long)g * N_ + n) * kp + c] = (__bf16)v;
}

// ---------------------------------------------------------------------------
// WMMA GEMM:  C[M,NC] = act(A[M,K](bf16) @ W[NCpad,K](bf16)^T + bias)
//
// block = 256 threads = 8 waves, block tile 128(M) x 128(N)
// wave tile = 64(M) x 32(N) = 8 accumulators of v_wmma_f32_16x16x32_bf16
//
// A tile (128 x 32 bf16 = 8KB) is double-buffered in LDS, filled with
// global_load_async_to_lds_b128 (2 per thread per stage, ASYNCcnt-tracked);
// the next stage is issued before waiting on the current one so the async
// DMA overlaps the 8-WMMA compute phase. LDS row stride is padded to 80B
// (20 banks) so the ds_load_b128 fragment reads are conflict-minimal.
//
// Steady-state inner loop: issue async stage(k+1), 4x global_load_b128 (B),
// 8x ds_load_b128 into distinct regs, ONE s_wait_dscnt, then 8 back-to-back
// v_wmma ops.
// ---------------------------------------------------------------------------
template<bool RELU, bool OUTBF16>
__global__ __launch_bounds__(256)
void gemm_wmma_kernel(const __bf16* __restrict__ A, int lda, long long strideAg,
                      const __bf16* __restrict__ W, int ldw,
                      const float* __restrict__ bias,
                      void* __restrict__ Cout, int ldc, long long strideCg,
                      int M, int NC, int K)
{
    __shared__ __align__(16) __bf16 lds_a[2][128][40];   // 40 elems = 80B stride

    int tid  = threadIdx.x;
    int g    = blockIdx.z;
    int wave = tid >> 5;
    int lane = tid & 31;
    int half = lane >> 4;
    int l16  = lane & 15;
    int rblk  = blockIdx.x * 128;
    int rwave = (wave >> 2) * 64;
    int cbase = blockIdx.y * 128 + (wave & 3) * 32;

    const __bf16* Ag = A + (long long)g * strideAg;
    char* Cg = (char*)Cout + (long long)g * strideCg * (OUTBF16 ? 2 : 4);

    // --- async staging assignment: 512 x 16B chunks per stage, 2 per thread
    int row0 = tid >> 2;            // rows 0..63
    int row1 = row0 + 64;           // rows 64..127
    int q    = tid & 3;             // 16B chunk within 64B k-row
    int ar0 = rblk + row0; if (ar0 > M - 1) ar0 = M - 1;
    int ar1 = rblk + row1; if (ar1 > M - 1) ar1 = M - 1;
    unsigned lbase = (unsigned)(size_t)&lds_a[0][0][0];  // LDS aperture: low 32 bits = byte offset
    unsigned lo0 = (unsigned)(row0 * 80 + q * 16);
    unsigned lo1 = (unsigned)(row1 * 80 + q * 16);
    const unsigned bufstride = 128 * 80;

    auto stage = [&](int buf, int k0) {
        const __bf16* g0 = Ag + (long long)ar0 * lda + k0 + q * 8;
        const __bf16* g1 = Ag + (long long)ar1 * lda + k0 + q * 8;
        unsigned boff = lbase + (unsigned)buf * bufstride;
        async_load_b128(boff + lo0, g0);
        async_load_b128(boff + lo1, g1);
    };

    v8f acc[4][2] = {};

    stage(0, 0);

    const __bf16* wp0 = W + (long long)(cbase + l16) * ldw + half * 8;
    const __bf16* wp1 = wp0 + (long long)16 * ldw;

    int nsteps = K >> 5;
    for (int it = 0; it < nsteps; ++it) {
        int k0  = it << 5;
        int cur = it & 1;
        bool more = (it + 1) < nsteps;
        if (more) stage(cur ^ 1, k0 + 32);     // overlap next-stage DMA

        // B fragments for this k-step (independent of LDS)
        union { v16bf v; i32x4 qd[2]; } b0, b1;
        b0.qd[0] = *(const i32x4*)(wp0 + k0);
        b0.qd[1] = *(const i32x4*)(wp0 + k0 + 16);
        b1.qd[0] = *(const i32x4*)(wp1 + k0);
        b1.qd[1] = *(const i32x4*)(wp1 + k0 + 16);

        if (more) wait_asynccnt<2>();          // current stage landed, next in flight
        else      wait_asynccnt<0>();
        __syncthreads();                       // all waves' async writes visible

        // load ALL four A fragments first (distinct regs) -> one DS wait,
        // then 8 back-to-back WMMAs
        union { v16bf v; i32x4 qd[2]; } a[4];
#pragma unroll
        for (int s = 0; s < 4; ++s) {
            const __bf16* lp = &lds_a[cur][rwave + s * 16 + l16][0];
            a[s].qd[0] = *(const i32x4*)(lp + half * 8);        // K {0..7}+8h
            a[s].qd[1] = *(const i32x4*)(lp + 16 + half * 8);   // K {16..23}+8h
        }
#pragma unroll
        for (int s = 0; s < 4; ++s) {
            acc[s][0] = __builtin_amdgcn_wmma_f32_16x16x32_bf16(
                            false, a[s].v, false, b0.v, (short)0, acc[s][0], false, false);
            acc[s][1] = __builtin_amdgcn_wmma_f32_16x16x32_bf16(
                            false, a[s].v, false, b1.v, (short)0, acc[s][1], false, false);
        }
        __syncthreads();                       // buffer free for stage it+2
    }

    // --- epilogue: bias + activation + guarded stores
#pragma unroll
    for (int n = 0; n < 2; ++n) {
        int col = cbase + n * 16 + l16;
        float bv = (col < NC) ? bias[col] : 0.f;
#pragma unroll
        for (int s = 0; s < 4; ++s) {
#pragma unroll
            for (int r = 0; r < 8; ++r) {
                int row = rblk + rwave + s * 16 + half * 8 + r;
                if (row < M && col < NC) {
                    float v = acc[s][n][r] + bv;
                    if (RELU) v = v > 0.f ? v : 0.f;
                    if (OUTBF16)
                        ((__bf16*)Cg)[(long long)row * ldc + col] = (__bf16)v;
                    else
                        ((float*)Cg)[(long long)row * ldc + col] = v;
                }
            }
        }
    }
}

// ---------------------------------------------------------------------------
// BN stats: per-channel sum / sumsq over N rows (also used for mean pooling)
// stats layout [G,128,2]
// ---------------------------------------------------------------------------
__global__ __launch_bounds__(128)
void bn_stats_kernel(const float* __restrict__ h, float* __restrict__ stats,
                     int rowsPerBlock)
{
    int g = blockIdx.z;
    int c = threadIdx.x;
    int r0 = blockIdx.x * rowsPerBlock;
    float s = 0.f, s2 = 0.f;
    for (int i = 0; i < rowsPerBlock; ++i) {
        float v = h[((long long)g * N_ + r0 + i) * 128 + c];
        s  += v;
        s2 += v * v;
    }
    unsafeAtomicAdd(&stats[(g * 128 + c) * 2 + 0], s);
    unsafeAtomicAdd(&stats[(g * 128 + c) * 2 + 1], s2);
}

// ---------------------------------------------------------------------------
// BN normalize (training-mode batch stats) + leaky relu -> xf
// ---------------------------------------------------------------------------
__global__ __launch_bounds__(256)
void bn_norm_kernel(const float* __restrict__ h, const float* __restrict__ stats,
                    const float* __restrict__ gamma, const float* __restrict__ beta,
                    float* __restrict__ xout)
{
    int g = blockIdx.z;
    long long t = (long long)blockIdx.x * blockDim.x + threadIdx.x;
    int n = (int)(t >> 7);
    int c = (int)(t & 127);
    if (n >= N_) return;
    float inv_n = 1.f / (float)N_;
    float mu  = stats[(g * 128 + c) * 2 + 0] * inv_n;
    float var = stats[(g * 128 + c) * 2 + 1] * inv_n - mu * mu;
    float v = (h[((long long)g * N_ + n) * 128 + c] - mu)
              * rsqrtf(var + 1e-5f) * gamma[c] + beta[c];
    v = v > 0.f ? v : 0.01f * v;
    xout[((long long)g * N_ + n) * 128 + c] = v;
}

// ---------------------------------------------------------------------------
// pooled_b[g,c] = bf16(sum[g,c] / N)
// ---------------------------------------------------------------------------
__global__ void pool_finish_kernel(const float* __restrict__ stats,
                                   __bf16* __restrict__ pooledb)
{
    int t = blockIdx.x * blockDim.x + threadIdx.x;
    if (t >= G_ * 128) return;
    pooledb[t] = (__bf16)(stats[t * 2] / (float)N_);
}

// ---------------------------------------------------------------------------
// Host launcher
// ---------------------------------------------------------------------------
extern "C" void kernel_launch(void* const* d_in, const int* in_sizes, int n_in,
                              void* d_out, int out_size, void* d_ws, size_t ws_size,
                              hipStream_t stream)
{
    (void)in_sizes; (void)n_in; (void)out_size; (void)ws_size;

    // ---- unpack inputs (setup_inputs dict order, params nested in order) ----
    int ii = 0;
    const float* nodef = (const float*)d_in[ii++];        // [G,N,16]
    const int*   fei   = (const int*)d_in[ii++];          // [G,2,E]
    const int*   bei   = (const int*)d_in[ii++];          // [G,2,E]
    const float* eattr = (const float*)d_in[ii++];        // [G,E,8]

    struct GP { const float *lw, *lb, *w1, *b1, *w2, *b2; };
    GP gp[2][3];
    for (int d = 0; d < 2; ++d)
        for (int l = 0; l < 3; ++l) {
            gp[d][l].lw = (const float*)d_in[ii++];
            gp[d][l].lb = (const float*)d_in[ii++];
            gp[d][l].w1 = (const float*)d_in[ii++];
            gp[d][l].b1 = (const float*)d_in[ii++];
            gp[d][l].w2 = (const float*)d_in[ii++];
            gp[d][l].b2 = (const float*)d_in[ii++];
        }
    const float *mw[3], *mb[3];
    for (int l = 0; l < 3; ++l) { mw[l] = (const float*)d_in[ii++]; mb[l] = (const float*)d_in[ii++]; }
    const float *bng[3], *bnb[3];
    for (int l = 0; l < 3; ++l) { bng[l] = (const float*)d_in[ii++]; bnb[l] = (const float*)d_in[ii++]; }
    const float *hw[3], *hb[3];
    for (int k = 0; k < 3; ++k) { hw[k] = (const float*)d_in[ii++]; hb[k] = (const float*)d_in[ii++]; }

    // ---- workspace carve-out ----
    char* ws = (char*)d_ws;
    size_t off = 0;
    auto take = [&](size_t bytes) -> char* {
        char* p = ws + off;
        off = (off + bytes + 255) & ~(size_t)255;
        return p;
    };
    float*  xf    = (float*) take((size_t)G_ * N_ * 128 * 4);
    float*  agg   = (float*) take((size_t)G_ * N_ * 128 * 4);
    __bf16* hin   = (__bf16*)take((size_t)G_ * N_ * 128 * 2);
    __bf16* y1    = (__bf16*)take((size_t)G_ * N_ * 128 * 2);
    __bf16* fxbx  = (__bf16*)take((size_t)G_ * N_ * 256 * 2);
    float*  stats = (float*) take((size_t)G_ * 128 * 2 * 4);
    __bf16* pooledb = (__bf16*)take((size_t)G_ * 128 * 2);
    __bf16* h1b   = (__bf16*)take((size_t)32 * 256 * 2);
    __bf16* h2b   = (__bf16*)take((size_t)32 * 128 * 2);

    __bf16 *w1b[2][3], *w2b[2][3], *mwb[3], *hwb[3];
    for (int d = 0; d < 2; ++d)
        for (int l = 0; l < 3; ++l) {
            int kp = (l == 0) ? 32 : 128;
            w1b[d][l] = (__bf16*)take((size_t)128 * kp * 2);
            w2b[d][l] = (__bf16*)take((size_t)128 * 128 * 2);
        }
    for (int l = 0; l < 3; ++l) mwb[l] = (__bf16*)take((size_t)128 * 256 * 2);
    hwb[0] = (__bf16*)take((size_t)256  * 128 * 2);
    hwb[1] = (__bf16*)take((size_t)128  * 256 * 2);
    hwb[2] = (__bf16*)take((size_t)1536 * 128 * 2);   // 1500 rows padded to 1536

    // ---- convert weights to bf16 (with padding) ----
    auto cvt = [&](const float* src, __bf16* dst, int rows, int cols,
                   int rowsPad, int colsPad) {
        long long total = (long long)rowsPad * colsPad;
        cvt_w_kernel<<<dim3((unsigned)((total + 255) / 256)), 256, 0, stream>>>(
            src, dst, rows, cols, rowsPad, colsPad);
    };
    for (int d = 0; d < 2; ++d)
        for (int l = 0; l < 3; ++l) {
            int din = (l == 0) ? 16 : 128;
            int kp  = (l == 0) ? 32 : 128;
            cvt(gp[d][l].w1, w1b[d][l], 128, din, 128, kp);
            cvt(gp[d][l].w2, w2b[d][l], 128, 128, 128, 128);
        }
    for (int l = 0; l < 3; ++l) cvt(mw[l], mwb[l], 128, 256, 128, 256);
    cvt(hw[0], hwb[0], 256, 128, 256, 128);
    cvt(hw[1], hwb[1], 128, 256, 128, 256);
    cvt(hw[2], hwb[2], 1500, 128, 1536, 128);

    // ---- GEMM dispatcher (block tile 128 x 128) ----
    auto gemm = [&](const __bf16* A, int lda, long long sAg,
                    const __bf16* W, int ldw, const float* bias,
                    void* C, int ldc, long long sCg,
                    int M, int NC, int K, int nz, bool relu, bool outbf) {
        dim3 grid((unsigned)((M + 127) / 128), (unsigned)((NC + 127) / 128), (unsigned)nz);
        if (relu) {
            if (outbf) gemm_wmma_kernel<true , true ><<<grid, 256, 0, stream>>>(A, lda, sAg, W, ldw, bias, C, ldc, sCg, M, NC, K);
            else       gemm_wmma_kernel<true , false><<<grid, 256, 0, stream>>>(A, lda, sAg, W, ldw, bias, C, ldc, sCg, M, NC, K);
        } else {
            if (outbf) gemm_wmma_kernel<false, true ><<<grid, 256, 0, stream>>>(A, lda, sAg, W, ldw, bias, C, ldc, sCg, M, NC, K);
            else       gemm_wmma_kernel<false, false><<<grid, 256, 0, stream>>>(A, lda, sAg, W, ldw, bias, C, ldc, sCg, M, NC, K);
        }
    };

    // ---- GNN layers ----
    const float* xcur = nodef;
    int ldx = 16;
    for (int l = 0; l < 3; ++l) {
        int din = (l == 0) ? 16 : 128;
        int kp  = (l == 0) ? 32 : 128;
        for (int d = 0; d < 2; ++d) {
            hipMemsetAsync(agg, 0, (size_t)G_ * N_ * 128 * 4, stream);
            const int* ei = d ? bei : fei;
            dim3 sg((unsigned)(((long long)E_ * din) / 256), 1, G_);
            scatter_gine_kernel<<<sg, 256, 0, stream>>>(
                xcur, ldx, (long long)N_ * ldx, ei, eattr,
                gp[d][l].lw, gp[d][l].lb, agg, din);
            dim3 pg((unsigned)(((long long)N_ * kp) / 256), 1, G_);
            prep_hin_kernel<<<pg, 256, 0, stream>>>(
                xcur, ldx, (long long)N_ * ldx, agg, hin, kp, din);
            // y1 = relu(hin @ w1^T + b1)
            gemm(hin, kp, (long long)N_ * kp, w1b[d][l], kp, gp[d][l].b1,
                 y1, 128, (long long)N_ * 128, N_, 128, kp, G_, true, true);
            // fx/bx = y1 @ w2^T + b2  -> fxbx columns [d*128, d*128+128)
            gemm(y1, 128, (long long)N_ * 128, w2b[d][l], 128, gp[d][l].b2,
                 fxbx + d * 128, 256, (long long)N_ * 256, N_, 128, 128, G_, false, true);
        }
        // merge: h = fxbx @ mw^T + mb  (f32, reuse agg buffer)
        gemm(fxbx, 256, (long long)N_ * 256, mwb[l], 256, mb[l],
             agg, 128, (long long)N_ * 128, N_, 128, 256, G_, false, false);
        // BN (training-mode batch stats) + leaky relu
        hipMemsetAsync(stats, 0, (size_t)G_ * 128 * 2 * 4, stream);
        bn_stats_kernel<<<dim3(128, 1, G_), 128, 0, stream>>>(agg, stats, N_ / 128);
        bn_norm_kernel<<<dim3((unsigned)((N_ * 128) / 256), 1, G_), 256, 0, stream>>>(
            agg, stats, bng[l], bnb[l], xf);
        xcur = xf;
        ldx = 128;
    }

    // ---- global mean pool ----
    hipMemsetAsync(stats, 0, (size_t)G_ * 128 * 2 * 4, stream);
    bn_stats_kernel<<<dim3(128, 1, G_), 128, 0, stream>>>(xf, stats, N_ / 128);
    pool_finish_kernel<<<dim3((G_ * 128 + 255) / 256), 256, 0, stream>>>(stats, pooledb);

    // ---- MLP head ----
    gemm(pooledb, 128, 0, hwb[0], 128, hb[0], h1b, 256, 0, 32, 256, 128, 1, true,  true);
    gemm(h1b,    256, 0, hwb[1], 256, hb[1], h2b, 128, 0, 32, 128, 256, 1, true,  true);
    gemm(h2b,    128, 0, hwb[2], 128, hb[2], d_out, 1500, 0, 32, 1500, 128, 1, false, false);
}
